// _MinGRULayer_34093450396042
// MI455X (gfx1250) — matrix operand order repaired and checked by reading, not compiled, not run on hardware
//
#include <hip/hip_runtime.h>
#include <hip/hip_bf16.h>
#include <math.h>

typedef __attribute__((ext_vector_type(16))) __bf16 v16bf;
typedef __attribute__((ext_vector_type(8)))  __bf16 v8bf;
typedef __attribute__((ext_vector_type(8)))  float  v8f;
typedef __attribute__((ext_vector_type(4)))  float  f4;
typedef __attribute__((ext_vector_type(4)))  int    i4v;
typedef i4v __attribute__((address_space(1)))* gaddr128_t;   // async src (global)
typedef i4v __attribute__((address_space(3)))* laddr128_t;   // async dst (LDS)

#define D_DIM   1024
#define T_DIM   4096
#define B_DIM   8
#define M_DIM   (B_DIM * T_DIM)   // 32768
#define N2_DIM  (2 * D_DIM)       // 2048
#define TILE_M  256
#define TILE_N  64                // d-columns per block (both halves handled)
#define TILE_K  32
#define NCHUNK  64
#define CT      64                // T per chunk
#define BD      (B_DIM * D_DIM)   // 8192
#define DQ      (D_DIM / 4)       // 256 float4s per row

// ---------- device math helpers ----------
__device__ __forceinline__ float softplusf(float x) {
    return fmaxf(x, 0.0f) + log1pf(expf(-fabsf(x)));
}
__device__ __forceinline__ float log_g(float x) {
    return (x >= 0.0f) ? logf(x + 0.5f) : -softplusf(-x);
}
__device__ __forceinline__ float logaddexpf_(float a, float b) {
    float m = fmaxf(a, b);
    if (m == -INFINITY) return -INFINITY;
    return m + log1pf(expf(-fabsf(a - b)));
}

// ---------- async global->LDS (CDNA5 path) with fallback ----------
#if __has_builtin(__builtin_amdgcn_global_load_async_to_lds_b128)
#define HAS_ASYNC_LDS 1
#else
#define HAS_ASYNC_LDS 0
#endif

__device__ __forceinline__ void wait_async0() {
#if __has_builtin(__builtin_amdgcn_s_wait_asynccnt)
    __builtin_amdgcn_s_wait_asynccnt(0);
#else
    asm volatile("s_wait_asynccnt 0x0" ::: "memory");
#endif
}

// Build a 16x32-layout bf16 A fragment from two contiguous 8-float runs in LDS.
__device__ __forceinline__ v16bf frag_from_f32(const float* run0, const float* run1) {
    const f4 x0 = *(const f4*)run0;
    const f4 x1 = *(const f4*)(run0 + 4);
    const f4 x2 = *(const f4*)run1;
    const f4 x3 = *(const f4*)(run1 + 4);
    v16bf r;
#pragma unroll
    for (int i = 0; i < 4; ++i) {
        r[i]      = (__bf16)x0[i];
        r[4 + i]  = (__bf16)x1[i];
        r[8 + i]  = (__bf16)x2[i];
        r[12 + i] = (__bf16)x3[i];
    }
    return r;
}

// ---------- Kernel 1: projection GEMM (bf16 WMMA) + fused gate math ----------
// projected = x @ W + b for column pair (d, d+1024), then
//   log_coeffs = -softplus(k), log_values = -softplus(-k) + log_g(h_tilde)
// Block: 256 threads (8 waves). Tile: 256 rows x 64 d-cols (x2 halves).
// Each wave owns two 16-row strips -> every B fragment feeds 2 WMMAs.
__global__ void __launch_bounds__(256)
mingru_proj_kernel(const float* __restrict__ x, const float* __restrict__ W,
                   const float* __restrict__ bias,
                   float* __restrict__ LC, float* __restrict__ LV) {
    __shared__ float  Asf[TILE_M][TILE_K];    // 256x32 f32, 32 KB (async target)
    __shared__ __bf16 Bh[TILE_N][TILE_K];     // [n][k] transposed, h-half, 4 KB
    __shared__ __bf16 Bk[TILE_N][TILE_K];     // [n][k] transposed, k-half, 4 KB

    const int m0   = blockIdx.x * TILE_M;
    const int n0   = blockIdx.y * TILE_N;
    const int tid  = threadIdx.x;
    const int lane = tid & 31;
    const int wave = tid >> 5;                // 0..7
    const int r    = lane & 15;
    const int hi   = lane >> 4;

    v8f acch[2][4], acck[2][4];
#pragma unroll
    for (int s = 0; s < 2; ++s)
#pragma unroll
        for (int nt = 0; nt < 4; ++nt) {
            acch[s][nt] = (v8f)(0.0f);
            acck[s][nt] = (v8f)(0.0f);
        }

    for (int k0 = 0; k0 < D_DIM; k0 += TILE_K) {
        __syncthreads();
        // --- A tile: 256 rows x 32 f32; thread t copies its row (8 x b128) ---
        {
            const float* srow = x + (size_t)(m0 + tid) * D_DIM + k0;
#if HAS_ASYNC_LDS
#pragma unroll
            for (int c = 0; c < 8; ++c)
                __builtin_amdgcn_global_load_async_to_lds_b128(
                    (gaddr128_t)(srow + 4 * c),
                    (laddr128_t)&Asf[tid][4 * c], 0, 0);
#else
#pragma unroll
            for (int c = 0; c < 8; ++c)
                *(f4*)&Asf[tid][4 * c] = *(const f4*)(srow + 4 * c);
#endif
        }
        // --- B tiles: W[k0..k0+31][n0..n0+63] (+1024) f32 -> bf16, transposed ---
        {
            const int kr = tid >> 3;
            const int c0 = (tid & 7) * 8;
            const float* sh = W + (size_t)(k0 + kr) * N2_DIM + n0 + c0;
            const float* sk = sh + D_DIM;
#pragma unroll
            for (int i = 0; i < 8; ++i) {
                Bh[c0 + i][kr] = (__bf16)sh[i];
                Bk[c0 + i][kr] = (__bf16)sk[i];
            }
            if (k0 + TILE_K < D_DIM)
                __builtin_prefetch(sh + TILE_K * N2_DIM);
        }
#if HAS_ASYNC_LDS
        wait_async0();
#endif
        __syncthreads();

        // --- A fragments (16x32 bf16, ISA layout), one per strip ---
        // lane<16: halves 0-7 -> K=0..7,  halves 8-15 -> K=16..23
        // lane>=16: halves 0-7 -> K=8..15, halves 8-15 -> K=24..31
        v16bf av[2];
#pragma unroll
        for (int s = 0; s < 2; ++s) {
            const int arow = 32 * wave + 16 * s + r;
            av[s] = frag_from_f32(&Asf[arow][8 * hi], &Asf[arow][16 + 8 * hi]);
        }

        // --- B fragments: lane = N, halves 0-7 -> K=8*hi.., halves 8-15 -> K=16+8*hi.. ---
        const int kb = 8 * hi;
#pragma unroll
        for (int nt = 0; nt < 4; ++nt) {
            const int brow = nt * 16 + r;
            const v8bf bh0 = *(const v8bf*)&Bh[brow][kb];
            const v8bf bh1 = *(const v8bf*)&Bh[brow][kb + 16];
            const v16bf bvh = __builtin_shufflevector(bh0, bh1,
                0, 1, 2, 3, 4, 5, 6, 7, 8, 9, 10, 11, 12, 13, 14, 15);
#pragma unroll
            for (int s = 0; s < 2; ++s)
                acch[s][nt] = __builtin_amdgcn_wmma_f32_16x16x32_bf16(
                    false, av[s], false, bvh, (short)0, acch[s][nt], false, false);

            const v8bf bk0 = *(const v8bf*)&Bk[brow][kb];
            const v8bf bk1 = *(const v8bf*)&Bk[brow][kb + 16];
            const v16bf bvk = __builtin_shufflevector(bk0, bk1,
                0, 1, 2, 3, 4, 5, 6, 7, 8, 9, 10, 11, 12, 13, 14, 15);
#pragma unroll
            for (int s = 0; s < 2; ++s)
                acck[s][nt] = __builtin_amdgcn_wmma_f32_16x16x32_bf16(
                    false, av[s], false, bvk, (short)0, acck[s][nt], false, false);
        }
    }

    // --- fused epilogue: bias + gate log-math, store log_coeffs / log_values ---
    // C/D layout: lane l, vgpr v -> M = v + 8*(l>=16), N = l%16
#pragma unroll
    for (int s = 0; s < 2; ++s) {
        const int mbase = m0 + 32 * wave + 16 * s;
#pragma unroll
        for (int nt = 0; nt < 4; ++nt) {
            const int d = n0 + nt * 16 + r;
            const float bh = bias[d];
            const float bk = bias[D_DIM + d];
#pragma unroll
            for (int v = 0; v < 8; ++v) {
                const int mm = mbase + v + 8 * hi;
                const float ph = acch[s][nt][v] + bh;   // h_tilde
                const float pk = acck[s][nt][v] + bk;   // gate pre-activation
                const float lc = -softplusf(pk);                  // log(1-z)
                const float lv = -softplusf(-pk) + log_g(ph);     // log(z)+log_g
                const size_t off = (size_t)mm * D_DIM + d;
                LC[off] = lc;
                LV[off] = lv;
            }
        }
    }
}

// ---------- Kernel 2a: per-chunk local scan summaries (float4) ----------
__global__ void __launch_bounds__(256)
mingru_scan_chunks_kernel(const f4* __restrict__ LC4, const f4* __restrict__ LV4,
                          f4* __restrict__ CA4, f4* __restrict__ CS4) {
    const int idx = blockIdx.x * blockDim.x + threadIdx.x; // 131072
    const int dq = idx & (DQ - 1);
    const int b  = (idx >> 8) & (B_DIM - 1);
    const int c  = idx >> 11;
    const size_t base = ((size_t)b * T_DIM + (size_t)c * CT) * DQ + dq;
    f4 A = (f4)(0.0f);
    f4 S = (f4)(-INFINITY);
    for (int t = 0; t < CT; ++t) {
        const f4 lc = LC4[base + (size_t)t * DQ];
        const f4 lv = LV4[base + (size_t)t * DQ];
#pragma unroll
        for (int j = 0; j < 4; ++j) {
            S[j] = logaddexpf_(lc[j] + S[j], lv[j]);
            A[j] += lc[j];
        }
    }
    const size_t o = (size_t)c * (BD / 4) + (size_t)b * DQ + dq;
    CA4[o] = A;
    CS4[o] = S;
}

// ---------- Kernel 2b: sequential scan over chunk summaries ----------
__global__ void __launch_bounds__(256)
mingru_scan_carries_kernel(const float* __restrict__ CA, const float* __restrict__ CS,
                           float* __restrict__ CR) {
    const int bd = blockIdx.x * blockDim.x + threadIdx.x;  // 8192
    float carry = -INFINITY;
    for (int c = 0; c < NCHUNK; ++c) {
        CR[(size_t)c * BD + bd] = carry;
        carry = logaddexpf_(CA[(size_t)c * BD + bd] + carry, CS[(size_t)c * BD + bd]);
    }
}

// ---------- Kernel 2c: re-scan with carry, h = exp(s), in-place over LC ----------
__global__ void __launch_bounds__(256)
mingru_scan_apply_kernel(f4* __restrict__ LC4, const f4* __restrict__ LV4,
                         const f4* __restrict__ CR4) {
    const int idx = blockIdx.x * blockDim.x + threadIdx.x;
    const int dq = idx & (DQ - 1);
    const int b  = (idx >> 8) & (B_DIM - 1);
    const int c  = idx >> 11;
    const size_t base = ((size_t)b * T_DIM + (size_t)c * CT) * DQ + dq;
    f4 s = CR4[(size_t)c * (BD / 4) + (size_t)b * DQ + dq];
    for (int t = 0; t < CT; ++t) {
        const size_t off = base + (size_t)t * DQ;
        const f4 lc = LC4[off];
        const f4 lv = LV4[off];
        f4 h;
#pragma unroll
        for (int j = 0; j < 4; ++j) {
            s[j] = logaddexpf_(lc[j] + s[j], lv[j]);
            h[j] = expf(s[j]);
        }
        LC4[off] = h;   // h overwrites log_coeffs
    }
}

// ---------- Kernel 3: residual + LayerNorm (float4) ----------
__global__ void __launch_bounds__(256)
mingru_ln_kernel(const f4* __restrict__ X4, const f4* __restrict__ H4,
                 const f4* __restrict__ G4, const f4* __restrict__ Be4,
                 f4* __restrict__ O4) {
    const int row = blockIdx.x;
    const int tid = threadIdx.x;
    const size_t base = (size_t)row * DQ;
    const f4 xv = X4[base + tid];
    const f4 hv = H4[base + tid];
    const f4 y = xv + hv;
    float s = y[0] + y[1] + y[2] + y[3];
    float ss = y[0] * y[0] + y[1] * y[1] + y[2] * y[2] + y[3] * y[3];
    __shared__ float r1[256];
    __shared__ float r2[256];
    r1[tid] = s; r2[tid] = ss;
    __syncthreads();
    for (int o = 128; o > 0; o >>= 1) {
        if (tid < o) { r1[tid] += r1[tid + o]; r2[tid] += r2[tid + o]; }
        __syncthreads();
    }
    const float mu  = r1[0] * (1.0f / D_DIM);
    const float var = r2[0] * (1.0f / D_DIM) - mu * mu;
    const float inv = rsqrtf(var + 1e-6f);
    const f4 g = G4[tid];
    const f4 be = Be4[tid];
    f4 o;
#pragma unroll
    for (int j = 0; j < 4; ++j)
        o[j] = (y[j] - mu) * inv * g[j] + be[j];
    O4[base + tid] = o;
}

extern "C" void kernel_launch(void* const* d_in, const int* in_sizes, int n_in,
                              void* d_out, int out_size, void* d_ws, size_t ws_size,
                              hipStream_t stream) {
    (void)in_sizes; (void)n_in; (void)out_size; (void)ws_size;
    const float* x     = (const float*)d_in[0];
    const float* W     = (const float*)d_in[1];
    const float* bias  = (const float*)d_in[2];
    const float* gamma = (const float*)d_in[3];
    const float* beta  = (const float*)d_in[4];

    float* ws = (float*)d_ws;
    const size_t MD = (size_t)M_DIM * D_DIM;          // 33,554,432 floats
    float* LC = ws;                                   // log_coeffs, later h
    float* LV = ws + MD;                              // log_values
    float* CA = ws + 2 * MD;                          // chunk A-sums   (64*8192)
    float* CS = CA + (size_t)NCHUNK * BD;             // chunk S-values
    float* CR = CS + (size_t)NCHUNK * BD;             // chunk carries

    dim3 g1(M_DIM / TILE_M, D_DIM / TILE_N);          // 128 x 16
    mingru_proj_kernel<<<g1, 256, 0, stream>>>(x, W, bias, LC, LV);

    const int scan_q = NCHUNK * BD / 4;               // 131072 threads
    mingru_scan_chunks_kernel<<<scan_q / 256, 256, 0, stream>>>(
        (const f4*)LC, (const f4*)LV, (f4*)CA, (f4*)CS);
    mingru_scan_carries_kernel<<<BD / 256, 256, 0, stream>>>(CA, CS, CR);
    mingru_scan_apply_kernel<<<scan_q / 256, 256, 0, stream>>>(
        (f4*)LC, (const f4*)LV, (const f4*)CR);

    mingru_ln_kernel<<<M_DIM, 256, 0, stream>>>(
        (const f4*)x, (const f4*)LC, (const f4*)gamma, (const f4*)beta, (f4*)d_out);
}
